// GATBase_51711406244277
// MI455X (gfx1250) — matrix operand order repaired and checked by reading, compiled, not run on hardware
//
#include <hip/hip_runtime.h>
#include <hip/hip_bf16.h>

// ---------------------------------------------------------------------------
// GAT forward for MI455X (gfx1250): wave32, WMMA bf16 (fp32 accum),
// per-layer precomputed fragment-ordered bf16 weights, async global->LDS
// staging (ASYNCcnt) when available, fused edge MLP, folded edge-attention.
// ---------------------------------------------------------------------------

typedef __attribute__((ext_vector_type(16))) __bf16 v16bf;
typedef __attribute__((ext_vector_type(8)))  __bf16 v8bf;
typedef __attribute__((ext_vector_type(8)))  float  v8f;
typedef int v4i __attribute__((__vector_size__(16)));

#define NN 100000
#define EE 1200000
#define NTILE_N (NN / 16)   // 6250 (exact)
#define NTILE_E (EE / 16)   // 75000 (exact)

// Fragment buffer layout (per layer), in bf16 elements:
#define FRAG_GW_OFF 0        // gat_W:   2 ksteps * 4 ntiles * 512 = 4096
#define FRAG_W1_OFF 4096     // eu_W1:   6 ksteps * 4 ntiles * 512 = 12288
#define FRAG_W2_OFF 16384    // eu_W2:   2 ksteps * 4 ntiles * 512 = 4096
#define FRAG_TOTAL  20480

#if defined(__has_builtin)
#  if __has_builtin(__builtin_amdgcn_global_load_async_to_lds_b128) && \
      __has_builtin(__builtin_amdgcn_s_wait_asynccnt)
#    define HAVE_ASYNC_LDS 1
#  endif
#endif
#ifndef HAVE_ASYNC_LDS
#  define HAVE_ASYNC_LDS 0
#endif

__device__ __forceinline__ float lrelu(float x) { return x > 0.f ? x : 0.2f * x; }

__device__ __forceinline__ v8f wmma_bf16(v16bf a, v16bf b, v8f c) {
  // D = A(16x32 bf16) * B(32x16 bf16) + C(16x16 f32)
  return __builtin_amdgcn_wmma_f32_16x16x32_bf16(false, a, false, b, (short)0, c,
                                                 false, false);
}

// Issue a contiguous bf16 copy global->LDS in 16B chunks. halves % 8 == 0.
__device__ __forceinline__ void stage_issue(__bf16* dst, const __bf16* src,
                                            int halves, int t, int nthr) {
#if HAVE_ASYNC_LDS
  for (int o = t * 8; o < halves; o += nthr * 8) {
    __builtin_amdgcn_global_load_async_to_lds_b128(
        (__attribute__((address_space(1))) v4i*)(unsigned long long)(src + o),
        (__attribute__((address_space(3))) v4i*)(unsigned)(unsigned long long)(dst + o),
        0, 0);
  }
#else
  for (int o = t * 8; o < halves; o += nthr * 8)
    *(v8bf*)(dst + o) = *(const v8bf*)(src + o);
#endif
}

__device__ __forceinline__ void stage_wait() {
#if HAVE_ASYNC_LDS
  __builtin_amdgcn_s_wait_asynccnt(0);
#endif
}

// ---------------------------------------------------------------------------
// Embedding encoders: sum of categorical embeddings.
// ---------------------------------------------------------------------------
__global__ void k_embed_node(const int* __restrict__ x, const float* __restrict__ at,
                             float* __restrict__ node) {
  long t = (long)blockIdx.x * blockDim.x + threadIdx.x;
  if (t >= (long)NN * 64) return;
  int n = (int)(t >> 6), d = (int)(t & 63);
  float s = 0.f;
#pragma unroll
  for (int f = 0; f < 9; ++f) {
    int xi = x[n * 9 + f];                   // [N,9], vocab 16
    s += at[((f << 4) + xi) * 64 + d];       // atom_tables [9,16,64]
  }
  node[t] = s;
}

__global__ void k_embed_edge(const int* __restrict__ ea, const float* __restrict__ bt,
                             float* __restrict__ edge) {
  long t = (long)blockIdx.x * blockDim.x + threadIdx.x;
  if (t >= (long)EE * 64) return;
  int e = (int)(t >> 6), d = (int)(t & 63);
  float s = 0.f;
#pragma unroll
  for (int f = 0; f < 3; ++f) {
    int xi = ea[e * 3 + f];                  // [E,3], vocab 8
    s += bt[((f << 3) + xi) * 64 + d];       // bond_tables [3,8,64]
  }
  edge[t] = s;
}

// ---------------------------------------------------------------------------
// Per-layer: convert gat_W / eu_W1 / eu_W2 to bf16 in WMMA B-fragment order.
// Done once; every GEMM block then stages a contiguous copy into LDS.
// ---------------------------------------------------------------------------
__global__ void k_prep_frag(const float* __restrict__ gatW,
                            const float* __restrict__ W1,
                            const float* __restrict__ W2, __bf16* __restrict__ frag) {
  int t = blockIdx.x * 256 + threadIdx.x;
  if (t >= FRAG_TOTAL) return;
  const float* src;
  int idx;
  if (t < FRAG_W1_OFF)      { src = gatW; idx = t; }
  else if (t < FRAG_W2_OFF) { src = W1;   idx = t - FRAG_W1_OFF; }
  else                      { src = W2;   idx = t - FRAG_W2_OFF; }
  int i = idx & 15, lane = (idx >> 4) & 31, fr = idx >> 9;
  int kk = fr >> 2, nt = fr & 3;
  int kr = kk * 32 + ((lane >> 4) << 3) + (i & 7) + ((i >> 3) << 4);   // lanes 16-31 hold K+8 (table order)
  int col = nt * 16 + (lane & 15);
  frag[t] = (__bf16)src[kr * 64 + col];
}

// ---------------------------------------------------------------------------
// Fold edgeproj into per-head attention vectors (eh only feeds a_edge):
//   w_att[h][d] = sum_c edgeproj_W[d][h*16+c] * att_edge[h][c]
// ---------------------------------------------------------------------------
__global__ void k_fold(const float* __restrict__ epW, const float* __restrict__ epb,
                       const float* __restrict__ ae, float* __restrict__ w_att,
                       float* __restrict__ b_att) {
  int t = threadIdx.x;                       // 256 threads, 1 block
  int h = t >> 6, d = t & 63;
  float s = 0.f;
#pragma unroll
  for (int c = 0; c < 16; ++c) s += epW[d * 64 + h * 16 + c] * ae[h * 16 + c];
  w_att[h * 64 + d] = s;
  if (t < 4) {
    float sb = 0.f;
#pragma unroll
    for (int c = 0; c < 16; ++c) sb += epb[t * 16 + c] * ae[t * 16 + c];
    b_att[t] = sb;
  }
}

// ---------------------------------------------------------------------------
// xh = node @ gat_W  (N x 64 @ 64 x 64), WMMA bf16.
// ---------------------------------------------------------------------------
__global__ void k_node_linear(const float* __restrict__ node,
                              const __bf16* __restrict__ gwf, float* __restrict__ xh) {
  __shared__ __align__(32) __bf16 sW[4096];
  int t = threadIdx.x;
  stage_issue(sW, gwf, 4096, t, 256);
  stage_wait();
  __syncthreads();

  int wave = t >> 5, lane = t & 31;
  int tile = blockIdx.x * 8 + wave;
  if (tile >= NTILE_N) return;                   // whole-wave uniform
  int arow = tile * 16 + (lane & 15);
  int koff = (lane >> 4) * 8;                    // A layout half-wave K offset

  v8f acc[4];
#pragma unroll
  for (int nt = 0; nt < 4; ++nt)
#pragma unroll
    for (int r = 0; r < 8; ++r) acc[nt][r] = 0.f;

#pragma unroll
  for (int kk = 0; kk < 2; ++kk) {
    const float* base = node + (long)arow * 64 + kk * 32;
    v16bf a;
#pragma unroll
    for (int i = 0; i < 8; ++i) {
      a[i]     = (__bf16)base[koff + i];
      a[i + 8] = (__bf16)base[koff + 16 + i];
    }
#pragma unroll
    for (int nt = 0; nt < 4; ++nt) {
      v16bf b = *reinterpret_cast<const v16bf*>(&sW[(kk * 4 + nt) * 512 + lane * 16]);
      acc[nt] = wmma_bf16(a, b, acc[nt]);
    }
  }
  int rbase = tile * 16 + ((lane >> 4) << 3);
#pragma unroll
  for (int nt = 0; nt < 4; ++nt) {
    int col = nt * 16 + (lane & 15);
#pragma unroll
    for (int r = 0; r < 8; ++r) xh[(long)(rbase + r) * 64 + col] = acc[nt][r];
  }
}

// a_src/a_dst per (node, head)
__global__ void k_attn_node(const float* __restrict__ xh, const float* __restrict__ asv,
                            const float* __restrict__ adv, float* __restrict__ a_src,
                            float* __restrict__ a_dst) {
  int t = blockIdx.x * blockDim.x + threadIdx.x;
  if (t >= NN * 4) return;
  int n = t >> 2, h = t & 3;
  float s = 0.f, d = 0.f;
#pragma unroll
  for (int c = 0; c < 16; ++c) {
    float v = xh[(long)n * 64 + h * 16 + c];
    s += v * asv[h * 16 + c];
    d += v * adv[h * 16 + c];
  }
  a_src[t] = s;
  a_dst[t] = d;
}

__global__ void k_init(float* __restrict__ agg, unsigned* __restrict__ m_enc,
                       float* __restrict__ den) {
  long t = (long)blockIdx.x * 256 + threadIdx.x;
  if (t >= (long)NN * 64) return;
  agg[t] = 0.f;
  if ((t & 63) < 4) {
    long idx = (t >> 6) * 4 + (t & 3);
    m_enc[idx] = ~0xFF800000u;  // order-preserving encoding of -inf
    den[idx] = 0.f;
  }
}

// logits + segment-max (uint-punned order-preserving atomicMax)
__global__ void k_logits(const float* __restrict__ edge, const int* __restrict__ srci,
                         const int* __restrict__ dsti, const float* __restrict__ a_src,
                         const float* __restrict__ a_dst, const float* __restrict__ w_att,
                         const float* __restrict__ b_att, float* __restrict__ logits,
                         unsigned* __restrict__ m_enc) {
  int e = blockIdx.x * blockDim.x + threadIdx.x;
  if (e >= EE) return;
  int s = srci[e], d = dsti[e];
  float er[64];
#pragma unroll
  for (int c = 0; c < 64; ++c) er[c] = edge[(long)e * 64 + c];
#pragma unroll
  for (int h = 0; h < 4; ++h) {
    float ae = b_att[h];
#pragma unroll
    for (int c = 0; c < 64; ++c) ae += er[c] * w_att[h * 64 + c];
    float lg = lrelu(a_src[s * 4 + h] + a_dst[d * 4 + h] + ae);
    logits[(long)e * 4 + h] = lg;
    unsigned bits = __float_as_uint(lg);
    unsigned enc = (bits & 0x80000000u) ? ~bits : (bits | 0x80000000u);
    atomicMax(&m_enc[d * 4 + h], enc);
  }
}

// ex = exp(logit - m[dst]); den[dst] += ex  (in-place into logits buffer)
__global__ void k_exp(const int* __restrict__ dsti, float* __restrict__ logits,
                      const unsigned* __restrict__ m_enc, float* __restrict__ den) {
  long t = (long)blockIdx.x * 256 + threadIdx.x;
  if (t >= (long)EE * 4) return;
  int e = (int)(t >> 2), h = (int)(t & 3);
  int d = dsti[e];
  unsigned enc = m_enc[d * 4 + h];
  unsigned bits = (enc & 0x80000000u) ? (enc & 0x7FFFFFFFu) : ~enc;
  float mm = __uint_as_float(bits);
  float ex = __expf(logits[t] - mm);
  logits[t] = ex;
  atomicAdd(&den[d * 4 + h], ex);
}

// agg[dst] += alpha * xh[src]  — one wave32 per edge, 2 channels per lane.
__global__ void k_scatter(const int* __restrict__ srci, const int* __restrict__ dsti,
                          const float* __restrict__ ex, const float* __restrict__ den,
                          const float* __restrict__ xh, float* __restrict__ agg) {
  int e = blockIdx.x * 8 + (threadIdx.x >> 5);
  if (e >= EE) return;
  int lane = threadIdx.x & 31;
  int s = srci[e], d = dsti[e];
#pragma unroll
  for (int half = 0; half < 2; ++half) {
    int j = lane + half * 32;
    int h = j >> 4;
    float alpha = ex[(long)e * 4 + h] / (den[d * 4 + h] + 1e-16f);
    atomicAdd(&agg[(long)d * 64 + j], alpha * xh[(long)s * 64 + j]);
  }
}

// conv = agg + b; node = relu(LN(conv)) + node   — one wave32 per node.
__global__ void k_node_post(const float* __restrict__ agg, const float* __restrict__ gb,
                            const float* __restrict__ g, const float* __restrict__ b,
                            float* __restrict__ node) {
  int warp = threadIdx.x >> 5, lane = threadIdx.x & 31;
  int n = blockIdx.x * 8 + warp;
  if (n >= NN) return;
  float v1 = agg[(long)n * 64 + lane] + gb[lane];
  float v2 = agg[(long)n * 64 + lane + 32] + gb[lane + 32];
  float s = v1 + v2;
#pragma unroll
  for (int m = 16; m >= 1; m >>= 1) s += __shfl_xor(s, m, 32);
  float mu = s * (1.f / 64.f);
  float d1 = v1 - mu, d2 = v2 - mu;
  float q = d1 * d1 + d2 * d2;
#pragma unroll
  for (int m = 16; m >= 1; m >>= 1) q += __shfl_xor(q, m, 32);
  float rs = rsqrtf(q * (1.f / 64.f) + 1e-5f);
  float y1 = d1 * rs * g[lane] + b[lane];
  float y2 = d2 * rs * g[lane + 32] + b[lane + 32];
  node[(long)n * 64 + lane] += fmaxf(y1, 0.f);
  node[(long)n * 64 + lane + 32] += fmaxf(y2, 0.f);
}

// ---------------------------------------------------------------------------
// Fused edge MLP: edge = relu(relu([h_src||h_dst||e] @ W1 + b1) @ W2 + b2)
// One wave per 16-edge tile; W1/W2 fragments async-staged into LDS;
// intermediate h re-laid-out C->A through per-wave LDS scratch (b128 ds ops).
// ---------------------------------------------------------------------------
__global__ void k_edge_mlp(const float* __restrict__ node, const int* __restrict__ srci,
                           const int* __restrict__ dsti, const __bf16* __restrict__ w1f,
                           const float* __restrict__ b1, const __bf16* __restrict__ w2f,
                           const float* __restrict__ b2, float* __restrict__ edge) {
  __shared__ __align__(32) __bf16 sB1[12288];  // 6 ksteps * 4 ntiles * 512
  __shared__ __align__(32) __bf16 sB2[4096];   // 2 ksteps * 4 ntiles * 512
  __shared__ __align__(32) __bf16 sH[8192];    // 8 waves * 16x64 staging
  __shared__ float sb1[64], sb2[64];
  int t = threadIdx.x;
  if (t == 0) __builtin_prefetch(w2f, 0, 1);   // global_prefetch_b8

  stage_issue(sB1, w1f, 12288, t, 256);
  stage_issue(sB2, w2f, 4096, t, 256);
  stage_wait();
  if (t < 64) { sb1[t] = b1[t]; sb2[t] = b2[t]; }
  __syncthreads();

  int wave = t >> 5, lane = t & 31;
  int tile = blockIdx.x * 8 + wave;            // E/16 tiles exactly
  int erow = tile * 16 + (lane & 15);
  int s = srci[erow], d = dsti[erow];
  int koff = (lane >> 4) * 8;
  const float* seg0 = node + (long)s * 64;
  const float* seg1 = node + (long)d * 64;
  const float* seg2 = edge + (long)erow * 64;

  v8f acc[4];
#pragma unroll
  for (int nt = 0; nt < 4; ++nt)
#pragma unroll
    for (int r = 0; r < 8; ++r) acc[nt][r] = 0.f;

#pragma unroll
  for (int kk = 0; kk < 6; ++kk) {             // K = 192 = 6 x 32
    const float* base = (kk < 2 ? seg0 : (kk < 4 ? seg1 : seg2)) + (kk & 1) * 32;
    v16bf a;
#pragma unroll
    for (int i = 0; i < 8; ++i) {
      a[i]     = (__bf16)base[koff + i];
      a[i + 8] = (__bf16)base[koff + 16 + i];
    }
#pragma unroll
    for (int nt = 0; nt < 4; ++nt) {
      v16bf bfr = *reinterpret_cast<const v16bf*>(&sB1[(kk * 4 + nt) * 512 + lane * 16]);
      acc[nt] = wmma_bf16(a, bfr, acc[nt]);
    }
  }

  // h = relu(acc + b1) -> per-wave LDS (C layout -> row-major)
  int hb = wave * 1024;
#pragma unroll
  for (int nt = 0; nt < 4; ++nt) {
    int col = nt * 16 + (lane & 15);
#pragma unroll
    for (int r = 0; r < 8; ++r) {
      int row = r + ((lane >> 4) << 3);
      sH[hb + row * 64 + col] = (__bf16)fmaxf(acc[nt][r] + sb1[col], 0.f);
    }
  }
  __builtin_amdgcn_wave_barrier();             // pin ds store->load order in wave

  v8f acc2[4];
#pragma unroll
  for (int nt = 0; nt < 4; ++nt)
#pragma unroll
    for (int r = 0; r < 8; ++r) acc2[nt][r] = 0.f;

#pragma unroll
  for (int kk = 0; kk < 2; ++kk) {             // K = 64 = 2 x 32
    const __bf16* hrow = sH + hb + (lane & 15) * 64 + kk * 32;
    v8bf lo = *reinterpret_cast<const v8bf*>(hrow + koff);        // ds_load_b128
    v8bf hi = *reinterpret_cast<const v8bf*>(hrow + koff + 16);   // ds_load_b128
    v16bf a;
#pragma unroll
    for (int i = 0; i < 8; ++i) { a[i] = lo[i]; a[i + 8] = hi[i]; }
#pragma unroll
    for (int nt = 0; nt < 4; ++nt) {
      v16bf bfr = *reinterpret_cast<const v16bf*>(&sB2[(kk * 4 + nt) * 512 + lane * 16]);
      acc2[nt] = wmma_bf16(a, bfr, acc2[nt]);
    }
  }
#pragma unroll
  for (int nt = 0; nt < 4; ++nt) {
    int col = nt * 16 + (lane & 15);
#pragma unroll
    for (int r = 0; r < 8; ++r) {
      int row = tile * 16 + r + ((lane >> 4) << 3);
      edge[(long)row * 64 + col] = fmaxf(acc2[nt][r] + sb2[col], 0.f);
    }
  }
}

// ---------------------------------------------------------------------------
extern "C" void kernel_launch(void* const* d_in, const int* in_sizes, int n_in,
                              void* d_out, int out_size, void* d_ws, size_t ws_size,
                              hipStream_t stream) {
  const int*   x        = (const int*)d_in[0];
  const int*   eattr    = (const int*)d_in[1];
  const int*   eidx     = (const int*)d_in[2];
  const float* at       = (const float*)d_in[3];
  const float* bt       = (const float*)d_in[4];
  const float* gat_W    = (const float*)d_in[5];
  const float* gat_b    = (const float*)d_in[6];
  const float* att_src  = (const float*)d_in[7];
  const float* att_dst  = (const float*)d_in[8];
  const float* att_edge = (const float*)d_in[9];
  const float* epW      = (const float*)d_in[10];
  const float* epb      = (const float*)d_in[11];
  const float* ln_g     = (const float*)d_in[12];
  const float* ln_b     = (const float*)d_in[13];
  const float* euW1     = (const float*)d_in[14];
  const float* eub1     = (const float*)d_in[15];
  const float* euW2     = (const float*)d_in[16];
  const float* eub2     = (const float*)d_in[17];

  float* node = (float*)d_out;                  // outputs double as live state
  float* edge = node + (long)NN * 64;
  const int* srci = eidx;
  const int* dsti = eidx + EE;

  float* ws = (float*)d_ws;
  float* xh    = ws;               ws += (long)NN * 64;
  float* a_src = ws;               ws += (long)NN * 4;
  float* a_dst = ws;               ws += (long)NN * 4;
  unsigned* m_enc = (unsigned*)ws; ws += (long)NN * 4;
  float* den   = ws;               ws += (long)NN * 4;
  float* agg   = ws;               ws += (long)NN * 64;
  float* exl   = ws;               ws += (long)EE * 4;
  float* w_att = ws;               ws += 256;
  float* b_att = ws;               ws += 4;
  __bf16* frag = (__bf16*)ws;      ws += FRAG_TOTAL / 2;   // 20480 bf16
  __bf16* gwf = frag + FRAG_GW_OFF;
  __bf16* w1f = frag + FRAG_W1_OFF;
  __bf16* w2f = frag + FRAG_W2_OFF;

  k_embed_node<<<(NN * 64 + 255) / 256, 256, 0, stream>>>(x, at, node);
  k_embed_edge<<<(int)(((long)EE * 64 + 255) / 256), 256, 0, stream>>>(eattr, bt, edge);

  for (int l = 0; l < 4; ++l) {
    k_prep_frag<<<(FRAG_TOTAL + 255) / 256, 256, 0, stream>>>(
        gat_W + l * 4096, euW1 + l * 192 * 64, euW2 + l * 4096, frag);
    k_fold<<<1, 256, 0, stream>>>(epW + l * 4096, epb + l * 64, att_edge + l * 64,
                                  w_att, b_att);
    k_node_linear<<<(NTILE_N + 7) / 8, 256, 0, stream>>>(node, gwf, xh);
    k_attn_node<<<(NN * 4 + 255) / 256, 256, 0, stream>>>(
        xh, att_src + l * 64, att_dst + l * 64, a_src, a_dst);
    k_init<<<(NN * 64 + 255) / 256, 256, 0, stream>>>(agg, m_enc, den);
    k_logits<<<(EE + 255) / 256, 256, 0, stream>>>(edge, srci, dsti, a_src, a_dst,
                                                   w_att, b_att, exl, m_enc);
    k_exp<<<(int)(((long)EE * 4 + 255) / 256), 256, 0, stream>>>(dsti, exl, m_enc, den);
    k_scatter<<<(EE + 7) / 8, 256, 0, stream>>>(srci, dsti, exl, den, xh, agg);
    k_node_post<<<NN / 8, 256, 0, stream>>>(agg, gat_b + l * 64, ln_g + l * 64,
                                            ln_b + l * 64, node);
    k_edge_mlp<<<NTILE_E / 8, 256, 0, stream>>>(node, srci, dsti, w1f,
                                                eub1 + l * 64, w2f,
                                                eub2 + l * 64, edge);
  }
}
